// FastSpeechModel_59639915872709
// MI455X (gfx1250) — matrix operand order repaired and checked by reading, not compiled, not run on hardware
//
#include <hip/hip_runtime.h>

// ---------------------------------------------------------------------------
// FastSpeech inference on MI455X (gfx1250).
// All matmul-shaped work -> v_wmma_f32_16x16x32_bf16 (wave32, f32 accum).
// Strides are template constants so B-side gathers compile to base+imm-offset
// load clauses (no per-element 64-bit address math, one wait per clause).
// Out-of-range rows/cols are clamped (feed only unstored outputs); K tails
// removed structurally (padded+zeroed score rows, V-buffer slack).
// One wave computes a 16x64 C strip: 1 A fragment feeds 4 WMMAs.
// ---------------------------------------------------------------------------

typedef __attribute__((ext_vector_type(16))) __bf16 v16bf;
typedef __attribute__((ext_vector_type(8)))  float  v8f;

constexpr int BB     = 16;    // batch
constexpr int TIN    = 128;   // input tokens
constexpr int DMODEL = 384;
constexpr int NH     = 6;
constexpr int DHEAD  = 64;
constexpr int DFF    = 1536;
constexpr int NLAYER = 4;
constexpr int OUTD   = 80;

#define ZERO8 {0.0f, 0.0f, 0.0f, 0.0f, 0.0f, 0.0f, 0.0f, 0.0f}

__device__ __forceinline__ v8f wmma_bf16(v16bf a, v16bf b, v8f c) {
  return __builtin_amdgcn_wmma_f32_16x16x32_bf16(false, a, false, b, (short)0, c,
                                                 false, false);
}

__device__ __forceinline__ void put4(v16bf& f, int base, float4 x, float s) {
  f[base + 0] = (__bf16)(x.x * s);
  f[base + 1] = (__bf16)(x.y * s);
  f[base + 2] = (__bf16)(x.z * s);
  f[base + 3] = (__bf16)(x.w * s);
}

// ---------------------------------------------------------------------------
// A fragment (16x32 bf16): lane&15 = row, hi = 16*(lane>>4);
// halves[0..15] = row[kb+hi .. +15] (contiguous k order, matches B).
// `rowk` points at row + kb + hi (16B aligned): 4x global_load_b128.
// ---------------------------------------------------------------------------
__device__ __forceinline__ v16bf frag_a_ld(const float* __restrict__ rowk,
                                           float scale) {
  const float4* p = (const float4*)rowk;
  float4 x0 = p[0], x1 = p[1], x2 = p[2], x3 = p[3];  // +0,+4,+8,+12 floats
  v16bf f;
  put4(f, 0, x0, scale);
  put4(f, 4, x1, scale);
  put4(f, 8, x2, scale);
  put4(f, 12, x3, scale);
  return f;
}

// B fragment (32x16 bf16), column gather with COMPILE-TIME stride LDB:
// pk points at element (k0, n); 16 loads at constant offsets i*LDB, then cvt.
template <int LDB>
__device__ __forceinline__ v16bf frag_bcol(const float* __restrict__ pk) {
  float t[16];
#pragma unroll
  for (int i = 0; i < 16; ++i) t[i] = pk[i * LDB];
  v16bf f;
#pragma unroll
  for (int i = 0; i < 16; ++i) f[i] = (__bf16)t[i];
  return f;
}

// B fragment from transposed [N][K] storage: 16 contiguous floats -> 4x b128.
__device__ __forceinline__ v16bf frag_brow(const float* __restrict__ rowk) {
  const float4* p = (const float4*)rowk;
  float4 x0 = p[0], x1 = p[1], x2 = p[2], x3 = p[3];
  v16bf f;
  put4(f, 0, x0, 1.0f);
  put4(f, 4, x1, 1.0f);
  put4(f, 8, x2, 1.0f);
  put4(f, 12, x3, 1.0f);
  return f;
}

// ---------------------------------------------------------------------------
// Generic GEMM: C[M,N] = A[M,K] @ B[K,N] + bias (+ res).
// A row stride = DMODEL (all call sites), B/C strides are template constants.
// K % 32 == 0.  One wave -> 16 rows x 64 cols.
// ---------------------------------------------------------------------------
template <int LDB, int LDC>
__global__ void gemm_wmma_kernel(const float* __restrict__ A,
                                 const float* __restrict__ Bm,
                                 const float* __restrict__ bias,
                                 const float* __restrict__ res,
                                 float* __restrict__ C,
                                 int M, int N, int K) {
  int tilesN = (N + 15) >> 4;
  int tilesM = (M + 15) >> 4;
  int g4 = (tilesN + 3) >> 2;
  int tid = blockIdx.x * (blockDim.x >> 5) + (threadIdx.x >> 5);
  if (tid >= tilesM * g4) return;
  int tm = tid / g4, tn0 = (tid - tm * g4) * 4;
  int lane = threadIdx.x & 31;
  int nn = lane & 15;
  int hi = (lane >> 4) << 4;
  int k16 = (lane >> 4) << 4;

  int ma = tm * 16 + nn;
  ma = ma < M ? ma : M - 1;                       // clamped A row
  const float* rowA = A + ma * DMODEL + hi;

  const float* pB[4];
#pragma unroll
  for (int t = 0; t < 4; ++t) {
    int n = (tn0 + t) * 16 + nn;
    n = n < N ? n : N - 1;                        // clamped B column
    pB[t] = Bm + k16 * LDB + n;
  }

  v8f acc[4] = {ZERO8, ZERO8, ZERO8, ZERO8};
  for (int kb = 0; kb < K; kb += 32) {
    v16bf af = frag_a_ld(rowA, 1.0f);
    rowA += 32;
#pragma unroll
    for (int t = 0; t < 4; ++t) {
      v16bf bf = frag_bcol<LDB>(pB[t]);
      pB[t] += 32 * LDB;
      acc[t] = wmma_bf16(af, bf, acc[t]);
    }
  }

  int mbase = tm * 16 + ((lane >> 4) << 3);
#pragma unroll
  for (int t = 0; t < 4; ++t) {
    int n = (tn0 + t) * 16 + nn;
    if (n >= N) continue;
    float bv = bias ? bias[n] : 0.0f;
#pragma unroll
    for (int j = 0; j < 8; ++j) {
      int m = mbase + j;
      if (m < M) {
        float v = acc[t][j] + bv;
        if (res) v += res[m * LDC + n];
        C[m * LDC + n] = v;
      }
    }
  }
}

// ---------------------------------------------------------------------------
// Conv1d (torch W[Cout][Cin][KT], 'SAME'): KT accumulating GEMMs with
// row-shifted A; zero padding folded into the A fragment scale.
// KT is a template constant -> W gather offsets i*KT are immediates.
// ---------------------------------------------------------------------------
template <int KT>
__global__ void conv_gemm_kernel(const float* __restrict__ X,
                                 const float* __restrict__ W,
                                 const float* __restrict__ bias,
                                 const float* __restrict__ res,
                                 float* __restrict__ Y,
                                 int T, int Cin, int Cout, int relu) {
  int M = BB * T;                    // multiple of 16 (BB == 16)
  int tilesN = Cout >> 4;            // Cout % 16 == 0
  int tilesM = M >> 4;
  int g4 = (tilesN + 3) >> 2;
  int tid = blockIdx.x * (blockDim.x >> 5) + (threadIdx.x >> 5);
  if (tid >= tilesM * g4) return;
  int tm = tid / g4, tn0 = (tid - tm * g4) * 4;
  int lane = threadIdx.x & 31;
  int nn = lane & 15;
  int hi = (lane >> 4) << 4;
  int k16 = (lane >> 4) << 4;
  int r = tm * 16 + nn;              // this lane's A row
  int bb = r / T, tt = r - bb * T;
  constexpr int pad = KT >> 1;

  const float* Wt[4];
#pragma unroll
  for (int t = 0; t < 4; ++t) {
    int n = (tn0 + t) * 16 + nn;
    n = n < Cout ? n : Cout - 1;
    Wt[t] = W + n * Cin * KT + k16 * KT;
  }

  v8f acc[4] = {ZERO8, ZERO8, ZERO8, ZERO8};
#pragma unroll
  for (int k = 0; k < KT; ++k) {
    int ts = tt + k - pad;
    int tsc = ts < 0 ? 0 : (ts >= T ? T - 1 : ts);
    float mz = (ts == tsc) ? 1.0f : 0.0f;        // zero-pad multiplier
    const float* rowX = X + (bb * T + tsc) * Cin + hi;
    const float* pW[4];
#pragma unroll
    for (int t = 0; t < 4; ++t) pW[t] = Wt[t] + k;
    for (int kb = 0; kb < Cin; kb += 32) {       // Cin % 32 == 0
      v16bf af = frag_a_ld(rowX, mz);
      rowX += 32;
#pragma unroll
      for (int t = 0; t < 4; ++t) {
        float tv[16];
#pragma unroll
        for (int i = 0; i < 16; ++i) tv[i] = pW[t][i * KT];
        v16bf bf;
#pragma unroll
        for (int i = 0; i < 16; ++i) bf[i] = (__bf16)tv[i];
        pW[t] += 32 * KT;
        acc[t] = wmma_bf16(af, bf, acc[t]);
      }
    }
  }

  int mbase = tm * 16 + ((lane >> 4) << 3);
#pragma unroll
  for (int t = 0; t < 4; ++t) {
    int n = (tn0 + t) * 16 + nn;
    if (n >= Cout) continue;
    float bv = bias[n];
#pragma unroll
    for (int j = 0; j < 8; ++j) {
      int m = mbase + j;
      float v = acc[t][j] + bv;
      if (res) v += res[m * Cout + n];
      if (relu) v = fmaxf(v, 0.0f);
      Y[m * Cout + n] = v;
    }
  }
}

// ---------------------------------------------------------------------------
// Attention scores per (b,h): S[:, :T] = (Q*0.125) @ K^T + mask, row stride Ts.
// K is read transposed from [B,T,DMODEL]: contiguous 16 floats per lane.
// ---------------------------------------------------------------------------
__global__ void attn_scores_kernel(const float* __restrict__ Q,
                                   const float* __restrict__ Kmat,
                                   const int* __restrict__ lens,
                                   float* __restrict__ S, int T, int Ts) {
  int bh = blockIdx.y;
  int b = bh / NH, h = bh - b * NH;
  int tiles = (T + 15) >> 4;
  int g4 = (tiles + 3) >> 2;
  int tid = blockIdx.x * (blockDim.x >> 5) + (threadIdx.x >> 5);
  if (tid >= tiles * g4) return;
  int tq = tid / g4, tk0 = (tid - tq * g4) * 4;
  const float* Qb = Q + (b * T) * DMODEL + h * DHEAD;
  const float* Kb = Kmat + (b * T) * DMODEL + h * DHEAD;
  int lane = threadIdx.x & 31;
  int nn = lane & 15;
  int hi = (lane >> 4) << 4;
  int k16 = (lane >> 4) << 4;

  int ma = tq * 16 + nn;
  ma = ma < T ? ma : T - 1;
  const float* rowQ = Qb + ma * DMODEL + hi;

  const float* krow[4];
#pragma unroll
  for (int t = 0; t < 4; ++t) {
    int n = (tk0 + t) * 16 + nn;
    n = n < T ? n : T - 1;
    krow[t] = Kb + n * DMODEL + k16;
  }

  v8f acc[4] = {ZERO8, ZERO8, ZERO8, ZERO8};
#pragma unroll
  for (int kb = 0; kb < DHEAD; kb += 32) {
    v16bf af = frag_a_ld(rowQ + kb, 0.125f);     // 1/sqrt(64)
#pragma unroll
    for (int t = 0; t < 4; ++t) {
      v16bf bf = frag_brow(krow[t] + kb);
      acc[t] = wmma_bf16(af, bf, acc[t]);
    }
  }

  float* S0 = S + (long long)bh * T * Ts;
  int mbase = tq * 16 + ((lane >> 4) << 3);
  int maxlen = lens[b];
#pragma unroll
  for (int t = 0; t < 4; ++t) {
    int n = (tk0 + t) * 16 + nn;
    if (n >= T) continue;
    float maskv = (n > maxlen) ? -1e30f : 0.0f;
#pragma unroll
    for (int j = 0; j < 8; ++j) {
      int m = mbase + j;
      if (m < T) S0[m * Ts + n] = acc[t][j] + maskv;
    }
  }
}

// ---------------------------------------------------------------------------
// ctx = softmax(S) @ V per (b,h).  K loop runs to Ts; softmax zeroed the pad
// (and V has 32 slack rows) so the tail contributes exactly zero.
// One wave: 16 rows x full head width (64).
// ---------------------------------------------------------------------------
__global__ void attn_ctx_kernel(const float* __restrict__ S,
                                const float* __restrict__ V,
                                float* __restrict__ O, int T, int Ts) {
  int bh = blockIdx.y;
  int b = bh / NH, h = bh - b * NH;
  int tilesM = (T + 15) >> 4;
  int tm = blockIdx.x * (blockDim.x >> 5) + (threadIdx.x >> 5);
  if (tm >= tilesM) return;
  const float* Sb = S + (long long)bh * T * Ts;
  const float* Vb = V + (b * T) * DMODEL + h * DHEAD;
  int lane = threadIdx.x & 31;
  int nn = lane & 15;
  int hi = (lane >> 4) << 4;
  int k16 = (lane >> 4) << 4;

  int ma = tm * 16 + nn;
  ma = ma < T ? ma : T - 1;
  const float* rowS = Sb + ma * Ts + hi;

  const float* pV[4];
#pragma unroll
  for (int t = 0; t < 4; ++t) pV[t] = Vb + k16 * DMODEL + t * 16 + nn;

  v8f acc[4] = {ZERO8, ZERO8, ZERO8, ZERO8};
  for (int kb = 0; kb < Ts; kb += 32) {
    v16bf af = frag_a_ld(rowS, 1.0f);
    rowS += 32;
#pragma unroll
    for (int t = 0; t < 4; ++t) {
      v16bf bf = frag_bcol<DMODEL>(pV[t]);
      pV[t] += 32 * DMODEL;
      acc[t] = wmma_bf16(af, bf, acc[t]);
    }
  }

  int mbase = tm * 16 + ((lane >> 4) << 3);
#pragma unroll
  for (int t = 0; t < 4; ++t) {
    int n = t * 16 + nn;
#pragma unroll
    for (int j = 0; j < 8; ++j) {
      int m = mbase + j;
      if (m < T)
        O[(b * T + m) * DMODEL + h * DHEAD + n] = acc[t][j];
    }
  }
}

// ---------------------------------------------------------------------------
// Softmax over [0,T), zeroes the pad [T,Ts).  One 256-thread block per row.
// ---------------------------------------------------------------------------
__global__ void softmax_kernel(float* __restrict__ S, int T, int Ts) {
  float* p = S + (long long)blockIdx.x * Ts;
  __shared__ float red[256];
  float mx = -3.4e38f;
  for (int i = threadIdx.x; i < T; i += 256) mx = fmaxf(mx, p[i]);
  red[threadIdx.x] = mx;
  __syncthreads();
  for (int s = 128; s > 0; s >>= 1) {
    if (threadIdx.x < (unsigned)s)
      red[threadIdx.x] = fmaxf(red[threadIdx.x], red[threadIdx.x + s]);
    __syncthreads();
  }
  mx = red[0];
  __syncthreads();
  float sum = 0.0f;
  for (int i = threadIdx.x; i < T; i += 256) {
    float e = __expf(p[i] - mx);
    p[i] = e;
    sum += e;
  }
  red[threadIdx.x] = sum;
  __syncthreads();
  for (int s = 128; s > 0; s >>= 1) {
    if (threadIdx.x < (unsigned)s) red[threadIdx.x] += red[threadIdx.x + s];
    __syncthreads();
  }
  float inv = 1.0f / red[0];
  for (int i = threadIdx.x; i < T; i += 256) p[i] *= inv;
  for (int i = T + threadIdx.x; i < Ts; i += 256) p[i] = 0.0f;  // zero pad
}

// LayerNorm over D=384, one row per 128-thread block, optional fused ReLU
__global__ void layernorm_kernel(const float* __restrict__ Yin,
                                 const float* __restrict__ g,
                                 const float* __restrict__ bta,
                                 float* __restrict__ Xout, int relu) {
  long long row = blockIdx.x;
  const float* y = Yin + row * DMODEL;
  float* x = Xout + row * DMODEL;
  __shared__ float s1[128], s2[128];
  float v0 = y[threadIdx.x], v1 = y[threadIdx.x + 128], v2 = y[threadIdx.x + 256];
  s1[threadIdx.x] = v0 + v1 + v2;
  s2[threadIdx.x] = v0 * v0 + v1 * v1 + v2 * v2;
  __syncthreads();
  for (int s = 64; s > 0; s >>= 1) {
    if (threadIdx.x < (unsigned)s) {
      s1[threadIdx.x] += s1[threadIdx.x + s];
      s2[threadIdx.x] += s2[threadIdx.x + s];
    }
    __syncthreads();
  }
  float mean = s1[0] * (1.0f / DMODEL);
  float var = s2[0] * (1.0f / DMODEL) - mean * mean;
  float inv = rsqrtf(var + 1e-5f);
  float vv[3] = {v0, v1, v2};
#pragma unroll
  for (int j = 0; j < 3; ++j) {
    int c = threadIdx.x + j * 128;
    float o = (vv[j] - mean) * inv * g[c] + bta[c];
    if (relu) o = fmaxf(o, 0.0f);
    x[c] = o;
  }
}

// x = 2*emb[tok] + pe(batch)  (reference quirk: pe indexed by batch)
__global__ void embed_kernel(const float* __restrict__ emb,
                             const int* __restrict__ tok,
                             float* __restrict__ X) {
  int i = blockIdx.x * blockDim.x + threadIdx.x;
  const int total = BB * TIN * DMODEL;
  if (i >= total) return;
  int d = i % DMODEL;
  int t = (i / DMODEL) % TIN;
  int b = i / (DMODEL * TIN);
  float e = emb[tok[b * TIN + t] * DMODEL + d];
  float den = __expf(-(float)(d & ~1) * (9.2103403720f / (float)DMODEL));
  float ang = (float)b * den;
  float pe = (d & 1) ? __cosf(ang) : __sinf(ang);
  X[i] = 2.0f * e + pe;
}

// length regulate: cumsum + searchsorted('right') gather, zero beyond length
__global__ void length_regulate_kernel(const float* __restrict__ Enc,
                                       const int* __restrict__ dur,
                                       float* __restrict__ Out, int Td) {
  int b = blockIdx.y;
  int p = blockIdx.x;
  __shared__ int cs[TIN];
  __shared__ int sidx, svalid;
  if (threadIdx.x == 0) {
    int s = 0;
    for (int j = 0; j < TIN; ++j) { s += dur[b * TIN + j]; cs[j] = s; }
    int lo = 0, hi = TIN;
    while (lo < hi) {
      int mid = (lo + hi) >> 1;
      if (cs[mid] <= p) lo = mid + 1; else hi = mid;
    }
    sidx = lo < TIN ? lo : TIN - 1;
    svalid = (p < cs[TIN - 1]) ? 1 : 0;
  }
  __syncthreads();
  const float* src = Enc + (b * TIN + sidx) * DMODEL;
  float* dst = Out + ((long long)b * Td + p) * DMODEL;
  for (int c = threadIdx.x; c < DMODEL; c += blockDim.x)
    dst[c] = svalid ? src[c] : 0.0f;
}

// duration predictor head: [B*TIN, 384] @ [384,1] + b
__global__ void dp_out_kernel(const float* __restrict__ Yv,
                              const float* __restrict__ w,
                              const float* __restrict__ b0,
                              float* __restrict__ out) {
  int m = blockIdx.x * blockDim.x + threadIdx.x;
  if (m >= BB * TIN) return;
  const float* row = Yv + m * DMODEL;
  float s = 0.0f;
  for (int c = 0; c < DMODEL; ++c) s += row[c] * w[c];
  out[m] = s + b0[0];
}

// [B*Td, OUT] -> [B, OUT, Td]
__global__ void transpose_out_kernel(const float* __restrict__ Yp,
                                     float* __restrict__ O, int Td) {
  long long i = (long long)blockIdx.x * blockDim.x + threadIdx.x;
  long long total = (long long)BB * OUTD * Td;
  if (i >= total) return;
  int t = (int)(i % Td);
  int o = (int)((i / Td) % OUTD);
  int b = (int)(i / ((long long)Td * OUTD));
  O[i] = Yp[((long long)b * Td + t) * OUTD + o];
}

// ---------------------------------------------------------------------------
// Host orchestration
// ---------------------------------------------------------------------------
static inline int cdiv_i(int a, int b) { return (a + b - 1) / b; }

struct TParams {
  const float *Wq, *bq, *Wk, *bk, *Wv, *bv, *Wo, *bo;
  const float *g1, *b1, *cw1, *cb1, *cw2, *cb2, *g2, *b2;
};

static dim3 ggrid(int M_, int N_) {
  int tilesM = (M_ + 15) / 16, g4 = ((N_ + 15) / 16 + 3) / 4;
  return dim3((unsigned)cdiv_i(tilesM * g4, 8));
}

static void run_layer(const TParams& P, int l, int T, const int* lens,
                      float* X, float* Y, float* Q, float* Kb, float* V,
                      float* A, float* Hb, float* S, hipStream_t stream) {
  const int M = BB * T;
  const int Ts = (T + 31) & ~31;
  dim3 blk(256);
  const float* Wq = P.Wq + (size_t)l * DMODEL * DMODEL;
  const float* Wk = P.Wk + (size_t)l * DMODEL * DMODEL;
  const float* Wv = P.Wv + (size_t)l * DMODEL * DMODEL;
  const float* Wo = P.Wo + (size_t)l * DMODEL * DMODEL;
  const float* bq = P.bq + (size_t)l * DMODEL;
  const float* bk = P.bk + (size_t)l * DMODEL;
  const float* bv = P.bv + (size_t)l * DMODEL;
  const float* bo = P.bo + (size_t)l * DMODEL;
  const float* g1 = P.g1 + (size_t)l * DMODEL;
  const float* b1 = P.b1 + (size_t)l * DMODEL;
  const float* cw1 = P.cw1 + (size_t)l * DFF * DMODEL * 9;
  const float* cb1 = P.cb1 + (size_t)l * DFF;
  const float* cw2 = P.cw2 + (size_t)l * DMODEL * DFF;
  const float* cb2 = P.cb2 + (size_t)l * DMODEL;
  const float* g2 = P.g2 + (size_t)l * DMODEL;
  const float* b2 = P.b2 + (size_t)l * DMODEL;

  gemm_wmma_kernel<DMODEL, DMODEL><<<ggrid(M, DMODEL), blk, 0, stream>>>(
      X, Wq, bq, nullptr, Q, M, DMODEL, DMODEL);
  gemm_wmma_kernel<DMODEL, DMODEL><<<ggrid(M, DMODEL), blk, 0, stream>>>(
      X, Wk, bk, nullptr, Kb, M, DMODEL, DMODEL);
  gemm_wmma_kernel<DMODEL, DMODEL><<<ggrid(M, DMODEL), blk, 0, stream>>>(
      X, Wv, bv, nullptr, V, M, DMODEL, DMODEL);

  int tilesT = cdiv_i(T, 16);
  int g4T = cdiv_i(tilesT, 4);
  attn_scores_kernel<<<dim3((unsigned)cdiv_i(tilesT * g4T, 8), BB * NH), blk, 0,
                       stream>>>(Q, Kb, lens, S, T, Ts);
  softmax_kernel<<<dim3((unsigned)(BB * NH * T)), blk, 0, stream>>>(S, T, Ts);
  attn_ctx_kernel<<<dim3((unsigned)cdiv_i(tilesT, 8), BB * NH), blk, 0,
                    stream>>>(S, V, A, T, Ts);

  // Y = A @ Wo + bo + X ; X = LN(Y)
  gemm_wmma_kernel<DMODEL, DMODEL><<<ggrid(M, DMODEL), blk, 0, stream>>>(
      A, Wo, bo, X, Y, M, DMODEL, DMODEL);
  layernorm_kernel<<<dim3((unsigned)M), dim3(128), 0, stream>>>(Y, g1, b1, X, 0);

  // FFN: H = relu(conv9(X)); Y = conv1(H) + X ; X = LN(Y)
  conv_gemm_kernel<9><<<ggrid(M, DFF), blk, 0, stream>>>(X, cw1, cb1, nullptr,
                                                         Hb, T, DMODEL, DFF, 1);
  conv_gemm_kernel<1><<<ggrid(M, DMODEL), blk, 0, stream>>>(Hb, cw2, cb2, X, Y,
                                                            T, DFF, DMODEL, 0);
  layernorm_kernel<<<dim3((unsigned)M), dim3(128), 0, stream>>>(Y, g2, b2, X, 0);
}

static TParams load_tp(void* const* d_in, int base) {
  TParams p;
  p.Wq = (const float*)d_in[base + 0];  p.bq = (const float*)d_in[base + 1];
  p.Wk = (const float*)d_in[base + 2];  p.bk = (const float*)d_in[base + 3];
  p.Wv = (const float*)d_in[base + 4];  p.bv = (const float*)d_in[base + 5];
  p.Wo = (const float*)d_in[base + 6];  p.bo = (const float*)d_in[base + 7];
  p.g1 = (const float*)d_in[base + 8];  p.b1 = (const float*)d_in[base + 9];
  p.cw1 = (const float*)d_in[base + 10]; p.cb1 = (const float*)d_in[base + 11];
  p.cw2 = (const float*)d_in[base + 12]; p.cb2 = (const float*)d_in[base + 13];
  p.g2 = (const float*)d_in[base + 14]; p.b2 = (const float*)d_in[base + 15];
  return p;
}

extern "C" void kernel_launch(void* const* d_in, const int* in_sizes, int n_in,
                              void* d_out, int out_size, void* d_ws,
                              size_t ws_size, hipStream_t stream) {
  (void)in_sizes; (void)n_in; (void)ws_size;

  // setup_inputs() insertion order:
  // 0: emb | 1..16: enc leaves | 17..32: dec leaves | 33..38: dp leaves
  // 39: out_w | 40: out_b | 41: tokens | 42: token_lengths
  // 43: melspec_length | 44: duration_multipliers | 45: t_out
  const float* emb = (const float*)d_in[0];
  TParams enc = load_tp(d_in, 1);
  TParams dec = load_tp(d_in, 17);
  const float* dp_cw = (const float*)d_in[33];
  const float* dp_cb = (const float*)d_in[34];
  const float* dp_g  = (const float*)d_in[35];
  const float* dp_b  = (const float*)d_in[36];
  const float* dp_ow = (const float*)d_in[37];
  const float* dp_ob = (const float*)d_in[38];
  const float* out_w = (const float*)d_in[39];
  const float* out_b = (const float*)d_in[40];
  const int* tokens  = (const int*)d_in[41];
  const int* tok_len = (const int*)d_in[42];
  const int* mel_len = (const int*)d_in[43];
  const int* dur     = (const int*)d_in[44];

  int Td = (out_size - BB * TIN) / (BB * OUTD);
  if (Td <= 0) return;

  // workspace layout (floats); V gets 32 rows of slack for the ctx K tail
  float* ws = (float*)d_ws;
  size_t MT = (size_t)BB * Td;
  float* X  = ws;
  float* Y  = X + MT * DMODEL;
  float* Q  = Y + MT * DMODEL;
  float* Kb = Q + MT * DMODEL;
  float* V  = Kb + MT * DMODEL;
  float* A  = V + (MT + 32) * DMODEL;    // +32-row slack after V
  float* Hb = A + MT * DMODEL;           // MT * DFF
  float* S  = Hb + MT * DFF;             // BB*NH*Td*roundup32(Td)

  dim3 blk(256);

  // ---------------- encoder ----------------
  embed_kernel<<<dim3((unsigned)cdiv_i(BB * TIN * DMODEL, 256)), blk, 0,
                 stream>>>(emb, tokens, X);
  for (int l = 0; l < NLAYER; ++l)
    run_layer(enc, l, TIN, tok_len, X, Y, Q, Kb, V, A, Hb, S, stream);

  // ---------------- duration predictor (reuses Q/Kb/V as temps) -------------
  const int ME = BB * TIN;
  conv_gemm_kernel<3><<<ggrid(ME, DMODEL), blk, 0, stream>>>(
      X, dp_cw, dp_cb, nullptr, Q, TIN, DMODEL, DMODEL, 0);
  layernorm_kernel<<<dim3((unsigned)ME), dim3(128), 0, stream>>>(Q, dp_g, dp_b,
                                                                 Kb, 1);
  conv_gemm_kernel<3><<<ggrid(ME, DMODEL), blk, 0, stream>>>(
      Kb, dp_cw + (size_t)DMODEL * DMODEL * 3, dp_cb + DMODEL, nullptr, V, TIN,
      DMODEL, DMODEL, 0);
  layernorm_kernel<<<dim3((unsigned)ME), dim3(128), 0, stream>>>(
      V, dp_g + DMODEL, dp_b + DMODEL, Q, 1);
  dp_out_kernel<<<dim3((unsigned)cdiv_i(ME, 256)), blk, 0, stream>>>(
      Q, dp_ow, dp_ob, (float*)d_out + (size_t)BB * OUTD * Td);

  // ---------------- length regulation: X(enc out) -> Y(dec in) ---------------
  length_regulate_kernel<<<dim3((unsigned)Td, BB), dim3(128), 0, stream>>>(
      X, dur, Y, Td);

  // ---------------- decoder (ping-pong: current activations live in Y) -------
  for (int l = 0; l < NLAYER; ++l)
    run_layer(dec, l, Td, mel_len, Y, X, Q, Kb, V, A, Hb, S, stream);

  // ---------------- output projection + transpose ----------------------------
  gemm_wmma_kernel<OUTD, OUTD><<<ggrid((int)MT, OUTD), blk, 0, stream>>>(
      Y, out_w, out_b, nullptr, Hb, (int)MT, OUTD, DMODEL);
  transpose_out_kernel<<<dim3((unsigned)cdiv_i(BB * OUTD * Td, 256)), blk, 0,
                         stream>>>(Hb, (float*)d_out, Td);
}
